// AttentionBlock_86715389706354
// MI455X (gfx1250) — compile-verified, matrix operand order checked by reading, not run on hardware
//
#include <hip/hip_runtime.h>
#include <stdint.h>

// Problem constants (match reference)
#define HB 512      // hidden
#define TB 2048     // sequence
#define BB 4        // batch
#define NHEADS 8
#define DH 64       // head dim

typedef _Float16 half_t;
typedef __attribute__((ext_vector_type(16))) _Float16 v16h;
typedef __attribute__((ext_vector_type(8)))  _Float16 v8h;
typedef __attribute__((ext_vector_type(8)))  float    v8f;
typedef __attribute__((ext_vector_type(4)))  float    v4f;

// Load a 16-element f16 fragment for WMMA A/B operands.
// Per ISA layout (16-bit A 16x32): lane group g (lane/16) holds K runs
// [g*8 .. g*8+7] and [16+g*8 .. 16+g*8+7] -> two contiguous 16-byte runs
// -> two b128 LDS/global loads.
__device__ __forceinline__ v16h load_frag(const half_t* row, int g) {
  v8h lo = *(const v8h*)(row + g * 8);
  v8h hi = *(const v8h*)(row + 16 + g * 8);
  v16h f;
#pragma unroll
  for (int i = 0; i < 8; ++i) { f[i] = lo[i]; f[i + 8] = hi[i]; }
  return f;
}

__device__ __forceinline__ v8f wmma16(v16h a, v16h b, v8f c) {
  return __builtin_amdgcn_wmma_f32_16x16x32_f16(
      /*neg_a=*/false, a, /*neg_b=*/false, b,
      /*c_mod=*/(short)0, c, /*reuse_a=*/false, /*reuse_b=*/false);
}

// Async DMA of one 16B segment from global (f16 data) into LDS, no VGPR
// round-trip. Tracked by ASYNCcnt. ldsOff = wave-relative LDS byte offset,
// gOff = byte offset from saddr base.
__device__ __forceinline__ void async_copy_b128(unsigned ldsOff, unsigned gOff,
                                                const half_t* base) {
  asm volatile("global_load_async_to_lds_b128 %0, %1, %2"
               :: "v"(ldsOff), "v"(gOff), "s"(base) : "memory");
}
__device__ __forceinline__ void wait_async0() {
  asm volatile("s_wait_asynccnt 0" ::: "memory");
}
__device__ __forceinline__ unsigned lds_off(const void* p) {
  return (unsigned)(uintptr_t)p;   // low 32 bits of flat shared addr = LDS offset
}

// ---------------------------------------------------------------------------
// Kernel 1: QKV projection. qkv = x @ w_qkv + b_qkv; Q pre-scaled by D^-0.5.
// Outputs Q/K/V in f16, head-major [b][h][t][d].
// Grid: (M/64 = 128, 3H/64 = 24), block 128 (4 waves).
// ---------------------------------------------------------------------------
__global__ __launch_bounds__(128) void qkv_gemm_kernel(
    const float* __restrict__ x, const float* __restrict__ w,
    const float* __restrict__ bias, half_t* __restrict__ Qh,
    half_t* __restrict__ Kh, half_t* __restrict__ Vh) {
  __shared__ __align__(16) half_t A_lds[64][40];   // [m][k], k=32 + pad
  __shared__ __align__(16) half_t Bt_lds[64][40];  // [n][k]

  const int mBase = blockIdx.x * 64;
  const int nBase = blockIdx.y * 64;
  const int tid  = threadIdx.x;
  const int wave = tid >> 5, lane = tid & 31;
  const int g = lane >> 4, idx = lane & 15;
  const int waveM = (wave >> 1) * 32, waveN = (wave & 1) * 32;
  const int ldw = 3 * HB;  // 1536

  v8f acc[2][2] = {};

  for (int k0 = 0; k0 < HB; k0 += 32) {
    // Stage A tile (f32 -> f16 conversion, VGPR path).
    {
      int row = tid >> 1, seg = tid & 1;
      const float* src = x + (size_t)(mBase + row) * HB + k0 + seg * 16;
      half_t* dst = &A_lds[row][seg * 16];
#pragma unroll
      for (int q2 = 0; q2 < 4; ++q2) {
        v4f xv = *(const v4f*)(src + q2 * 4);
#pragma unroll
        for (int c = 0; c < 4; ++c) dst[q2 * 4 + c] = (half_t)xv[c];
      }
    }
    // Stage B^T tile: Bt[n][k] = w[k0+k][nBase+n]  (f32 -> f16)
    {
      int n = tid >> 1, kh = (tid & 1) * 16;
#pragma unroll
      for (int kk = 0; kk < 16; ++kk)
        Bt_lds[n][kh + kk] = (half_t)w[(size_t)(k0 + kh + kk) * ldw + nBase + n];
    }
    __syncthreads();

    v16h af[2], bf[2];
#pragma unroll
    for (int i = 0; i < 2; ++i)
      af[i] = load_frag(&A_lds[waveM + i * 16 + idx][0], g);
#pragma unroll
    for (int j = 0; j < 2; ++j)
      bf[j] = load_frag(&Bt_lds[waveN + j * 16 + idx][0], g);
#pragma unroll
    for (int i = 0; i < 2; ++i)
#pragma unroll
      for (int j = 0; j < 2; ++j)
        acc[i][j] = wmma16(af[i], bf[j], acc[i][j]);
    __syncthreads();
  }

  // Epilogue: bias, Q scaling, scatter to head-major f16 buffers.
  const int which = nBase / HB;              // 0=Q 1=K 2=V (tile never straddles)
  const int head  = (nBase % HB) / DH;       // tile never straddles a head
  half_t* dstBuf = (which == 0) ? Qh : (which == 1) ? Kh : Vh;
  const float scale = (which == 0) ? 0.125f : 1.0f;  // D^-0.5 = 1/8
#pragma unroll
  for (int j = 0; j < 2; ++j) {
    const int dcol = waveN + j * 16 + idx;   // d within head (0..63)
    const float bv = bias[nBase + dcol];
#pragma unroll
    for (int i = 0; i < 2; ++i) {
#pragma unroll
      for (int r = 0; r < 8; ++r) {
        int mg = mBase + waveM + i * 16 + r + g * 8;  // D-layout row
        int bi = mg >> 11;                             // / T
        int t  = mg & (TB - 1);
        float v = (acc[i][j][r] + bv) * scale;
        dstBuf[(((size_t)bi * NHEADS + head) * TB + t) * DH + dcol] = (half_t)v;
      }
    }
  }
}

// ---------------------------------------------------------------------------
// Kernel 2: flash attention per (b,h). 64 query rows per block, key blocks of
// 64, online softmax in WMMA D-layout. K block staged into LDS once per block
// via async DMA (shared by all 4 waves). Grid: (B*H=32, T/64=32), block 128.
// ---------------------------------------------------------------------------
__global__ __launch_bounds__(128) void attn_kernel(
    const half_t* __restrict__ Qh, const half_t* __restrict__ Kh,
    const half_t* __restrict__ Vh, half_t* __restrict__ Ah) {
  __shared__ __align__(16) half_t Klds[64][72];     // K block, row-major + pad
  __shared__ __align__(16) half_t Vt[64][72];       // V^T: [d][key] + pad
  __shared__ __align__(16) half_t Plds[4][16][72];  // per-wave P (16x64 + pad)

  const int bh = blockIdx.x;
  const int b = bh / NHEADS, h = bh % NHEADS;
  const int qBase = blockIdx.y * 64;
  const int tid  = threadIdx.x;
  const int wave = tid >> 5, lane = tid & 31;
  const int g = lane >> 4, idx = lane & 15;

  const half_t* Qp = Qh + (size_t)bh * TB * DH;
  const half_t* Kp = Kh + (size_t)bh * TB * DH;
  const half_t* Vp = Vh + (size_t)bh * TB * DH;

  // Q fragments (A operand), resident for the whole kernel.
  v16h qf[2];
#pragma unroll
  for (int ks = 0; ks < 2; ++ks)
    qf[ks] = load_frag(Qp + (size_t)(qBase + wave * 16 + idx) * DH + ks * 32, g);

  v8f o[4] = {};
  float rowMax[8], rowSum[8];
#pragma unroll
  for (int r = 0; r < 8; ++r) { rowMax[r] = -3.0e38f; rowSum[r] = 0.f; }

  for (int j0 = 0; j0 < TB; j0 += 64) {
    __syncthreads();  // prior-iteration consumers of Klds/Vt are done

    // --- Stage K block via async DMA: 64 rows x 128B = 512 x 16B segments.
#pragma unroll
    for (int it = 0; it < 4; ++it) {
      int s = tid + it * 128;
      int row = s >> 3, c = (s & 7) * 8;  // halves
      async_copy_b128(lds_off(&Klds[row][c]),
                      (unsigned)(((j0 + row) * DH + c) * sizeof(half_t)), Kp);
    }
    // --- Stage V^T via VALU (transpose): Vt[d][key] = V[j0+key][d]
    {
      int key = tid >> 1, dbase = (tid & 1) * 32;
      const half_t* vrow = Vp + (size_t)(j0 + key) * DH + dbase;
#pragma unroll
      for (int i = 0; i < 32; ++i) Vt[dbase + i][key] = vrow[i];
    }
    if (j0 + 64 < TB) {  // pull next K/V block toward L2
      __builtin_prefetch(Kp + (size_t)(j0 + 64 + lane) * DH, 0, 0);
      __builtin_prefetch(Vp + (size_t)(j0 + 64 + lane) * DH, 0, 0);
    }
    wait_async0();       // this wave's async LDS writes landed
    __syncthreads();     // all waves' staging visible

    // --- S = Q K^T from LDS-resident K block.
    v8f s[4];
#pragma unroll
    for (int nt = 0; nt < 4; ++nt) {
      v8f a = {};
#pragma unroll
      for (int ks = 0; ks < 2; ++ks) {
        v16h bf = load_frag(&Klds[nt * 16 + idx][ks * 32], g);
        a = wmma16(qf[ks], bf, a);
      }
      s[nt] = a;
    }
    // --- Online softmax. Row m = r + g*8 lives across the 16 lanes of group g.
#pragma unroll
    for (int r = 0; r < 8; ++r) {
      float mx = s[0][r];
#pragma unroll
      for (int nt = 1; nt < 4; ++nt) mx = fmaxf(mx, s[nt][r]);
#pragma unroll
      for (int off = 1; off < 16; off <<= 1)
        mx = fmaxf(mx, __shfl_xor(mx, off, 32));
      float mnew = fmaxf(rowMax[r], mx);
      float corr = __expf(rowMax[r] - mnew);
      float lsum = 0.f;
#pragma unroll
      for (int nt = 0; nt < 4; ++nt) {
        float p = __expf(s[nt][r] - mnew);
        s[nt][r] = p;
        lsum += p;
        o[nt][r] *= corr;
      }
#pragma unroll
      for (int off = 1; off < 16; off <<= 1)
        lsum += __shfl_xor(lsum, off, 32);
      rowSum[r] = rowSum[r] * corr + lsum;
      rowMax[r] = mnew;
    }

    // --- P tile: D-layout -> row-major LDS (same-wave producer/consumer,
    // DS ops are in-order per wave -> no barrier needed).
#pragma unroll
    for (int nt = 0; nt < 4; ++nt)
#pragma unroll
      for (int r = 0; r < 8; ++r)
        Plds[wave][r + g * 8][nt * 16 + idx] = (half_t)s[nt][r];

    // --- O += P @ V
    v16h pa[2];
#pragma unroll
    for (int ks = 0; ks < 2; ++ks)
      pa[ks] = load_frag(&Plds[wave][idx][ks * 32], g);
#pragma unroll
    for (int nt = 0; nt < 4; ++nt) {
#pragma unroll
      for (int ks = 0; ks < 2; ++ks) {
        v16h vb = load_frag(&Vt[nt * 16 + idx][ks * 32], g);
        o[nt] = wmma16(pa[ks], vb, o[nt]);
      }
    }
  }

  // Normalize and store attn output as f16, layout [b][t][h*D+d] for proj GEMM.
#pragma unroll
  for (int nt = 0; nt < 4; ++nt) {
#pragma unroll
    for (int r = 0; r < 8; ++r) {
      float v = o[nt][r] / rowSum[r];
      int t = qBase + wave * 16 + r + g * 8;
      int d = nt * 16 + idx;
      Ah[((size_t)b * TB + t) * HB + h * DH + d] = (half_t)v;
    }
  }
}

// ---------------------------------------------------------------------------
// Kernel 3: output projection. out = attn @ w_proj + b_proj (f32 out).
// A tile (f16) staged via async DMA. Grid: (M/64 = 128, H/64 = 8), block 128.
// ---------------------------------------------------------------------------
__global__ __launch_bounds__(128) void proj_gemm_kernel(
    const half_t* __restrict__ Ah, const float* __restrict__ w,
    const float* __restrict__ bias, float* __restrict__ out) {
  __shared__ __align__(16) half_t A_lds[64][40];
  __shared__ __align__(16) half_t Bt_lds[64][40];

  const int mBase = blockIdx.x * 64;
  const int nBase = blockIdx.y * 64;
  const int tid  = threadIdx.x;
  const int wave = tid >> 5, lane = tid & 31;
  const int g = lane >> 4, idx = lane & 15;
  const int waveM = (wave >> 1) * 32, waveN = (wave & 1) * 32;

  v8f acc[2][2] = {};

  for (int k0 = 0; k0 < HB; k0 += 32) {
    // A tile: 64 rows x 64B = 256 x 16B segments, async DMA (f16 source).
#pragma unroll
    for (int it = 0; it < 2; ++it) {
      int s = tid + it * 128;
      int row = s >> 2, c = (s & 3) * 8;  // halves
      async_copy_b128(lds_off(&A_lds[row][c]),
                      (unsigned)((((size_t)(mBase + row) * HB + k0 + c)) *
                                 sizeof(half_t)),
                      Ah);
    }
    // B^T tile from w_proj (f32 [H][H]) with conversion.
    {
      int n = tid >> 1, kh = (tid & 1) * 16;
#pragma unroll
      for (int kk = 0; kk < 16; ++kk)
        Bt_lds[n][kh + kk] = (half_t)w[(size_t)(k0 + kh + kk) * HB + nBase + n];
    }
    wait_async0();
    __syncthreads();

    v16h af[2], bf[2];
#pragma unroll
    for (int i = 0; i < 2; ++i)
      af[i] = load_frag(&A_lds[waveM + i * 16 + idx][0], g);
#pragma unroll
    for (int j = 0; j < 2; ++j)
      bf[j] = load_frag(&Bt_lds[waveN + j * 16 + idx][0], g);
#pragma unroll
    for (int i = 0; i < 2; ++i)
#pragma unroll
      for (int j = 0; j < 2; ++j)
        acc[i][j] = wmma16(af[i], bf[j], acc[i][j]);
    __syncthreads();
  }

#pragma unroll
  for (int j = 0; j < 2; ++j) {
    int ng = nBase + waveN + j * 16 + idx;
    float bv = bias[ng];
#pragma unroll
    for (int i = 0; i < 2; ++i)
#pragma unroll
      for (int r = 0; r < 8; ++r) {
        int mg = mBase + waveM + i * 16 + r + g * 8;
        out[(size_t)mg * HB + ng] = acc[i][j][r] + bv;
      }
  }
}

// ---------------------------------------------------------------------------
extern "C" void kernel_launch(void* const* d_in, const int* in_sizes, int n_in,
                              void* d_out, int out_size, void* d_ws, size_t ws_size,
                              hipStream_t stream) {
  const float* x      = (const float*)d_in[0];
  const float* w_qkv  = (const float*)d_in[1];
  const float* b_qkv  = (const float*)d_in[2];
  const float* w_proj = (const float*)d_in[3];
  const float* b_proj = (const float*)d_in[4];
  float* out = (float*)d_out;

  const size_t nElem = (size_t)BB * TB * HB;  // 4M elements
  half_t* Qh = (half_t*)d_ws;                 // f16, head-major
  half_t* Kh = Qh + nElem;
  half_t* Vh = Kh + nElem;
  half_t* Ah = Vh + nElem;                    // attn output, [b][t][h*D+d]
  // total scratch: 4 * 4M * 2B = 32 MB

  dim3 blk(128);
  qkv_gemm_kernel<<<dim3((BB * TB) / 64, (3 * HB) / 64), blk, 0, stream>>>(
      x, w_qkv, b_qkv, Qh, Kh, Vh);
  attn_kernel<<<dim3(BB * NHEADS, TB / 64), blk, 0, stream>>>(Qh, Kh, Vh, Ah);
  proj_gemm_kernel<<<dim3((BB * TB) / 64, HB / 64), blk, 0, stream>>>(
      Ah, w_proj, b_proj, out);
}
